// LSTMCRF_50878182588883
// MI455X (gfx1250) — matrix operand order repaired and checked by reading, not compiled, not run on hardware
//
#include <hip/hip_runtime.h>
#include <hip/hip_bf16.h>
#include <math.h>

// ---------------------------------------------------------------------------
// BiLSTM + CRF NLL for MI455X (gfx1250), wave32, WMMA bf16 -> f32 accumulate.
// E=256 H=512 V=50000 T=25 B=32 S=256
// ---------------------------------------------------------------------------

typedef __bf16 bf16;
typedef __attribute__((ext_vector_type(16))) __bf16 v16bf;
typedef __attribute__((ext_vector_type(8)))  __bf16 v8bf;
typedef __attribute__((ext_vector_type(8)))  float  v8f;

#define E_DIM 256
#define H_DIM 512
#define G_DIM 2048      // 4*H
#define T_TAGS 25
#define BATCH 32
#define SEQ 256
#define ROWS (SEQ*BATCH)  // 8192

// LDS strides (elements), padded for bank-conflict-free fragment loads
#define HSTR 520    // bf16 row stride for h (16 rows)
#define GSTR 2052   // f32 row stride for gate staging (16 rows)
#define LSTM_LDS_BYTES (16*HSTR*2 + 16*GSTR*4)   // 147,968 B < 320 KB/WGP

// ---- WMMA fragment loaders ------------------------------------------------
// A-matrix 16x32 bf16: lane<16 holds K=k0..k0+15,
// lane>=16 holds K=k0+16..k0+31; row = lane%16. Two 16B loads per lane.
__device__ __forceinline__ v16bf load_a_frag(const bf16* __restrict__ p, int ld, int lane) {
  const int half = lane >> 4, r = lane & 15;
  const bf16* q = p + r * ld + 16 * half;
  v8bf lo = *(const v8bf*)(q);
  v8bf hi = *(const v8bf*)(q + 8);
  return __builtin_shufflevector(lo, hi, 0,1,2,3,4,5,6,7,8,9,10,11,12,13,14,15);
}
// B-matrix 32x16 bf16 (contiguous k, same order as A): lanes 0-15 hold K=k0..k0+15 for
// column n=lane, lanes 16-31 hold K=k0+16..k0+31. p is B^T (N x K row-major),
// so one contiguous 32B load per lane.
__device__ __forceinline__ v16bf load_b_frag(const bf16* __restrict__ p, int ld, int lane) {
  const int half = lane >> 4, r = lane & 15;
  return *(const v16bf*)(p + r * ld + 16 * half);
}
__device__ __forceinline__ v8f wmma_bf16(v16bf a, v16bf b, v8f c) {
  return __builtin_amdgcn_wmma_f32_16x16x32_bf16(false, a, false, b, (short)0, c, false, false);
}

__device__ __forceinline__ float sigmoidf_(float x) { return 1.0f / (1.0f + __expf(-x)); }

// ---- f32 -> bf16 weight conversion ----------------------------------------
__global__ void cvt_bf16_kernel(const float* __restrict__ src, bf16* __restrict__ dst, int n) {
  for (int i = blockIdx.x * blockDim.x + threadIdx.x; i < n; i += gridDim.x * blockDim.x)
    dst[i] = (bf16)src[i];
}

// ---- embedding gather, time-major rows (row = s*32 + b), bf16 --------------
__global__ void gather_kernel(const float* __restrict__ tab, const int* __restrict__ ix,
                              bf16* __restrict__ x) {
  int tid = blockIdx.x * blockDim.x + threadIdx.x;
  if (tid >= ROWS * E_DIM) return;
  int e = tid & (E_DIM - 1), row = tid >> 8;
  int b = row & 31, s = row >> 5;
  int v = ix[b * SEQ + s];
  x[tid] = (bf16)tab[(size_t)v * E_DIM + e];
}

// ---- input projection: xp = X(8192x256) @ W^T(256x2048) + b_ih  (both dirs)
// block = 256 threads (8 waves); wave -> 16(M) x 64(N) strip, K loop of 8.
__global__ void inproj_kernel(const bf16* __restrict__ x,
                              const bf16* __restrict__ w_f, const bf16* __restrict__ w_b,
                              const float* __restrict__ bih_f, const float* __restrict__ bih_b,
                              float* __restrict__ xp_f, float* __restrict__ xp_b) {
  const int dir = blockIdx.z;
  const bf16*  w   = dir ? w_b   : w_f;
  const float* bih = dir ? bih_b : bih_f;
  float*       xp  = dir ? xp_b  : xp_f;
  const int lane = threadIdx.x & 31;
  const int wave = threadIdx.x >> 5;
  const int m0 = blockIdx.y * 16;
  const int n0 = blockIdx.x * 512 + wave * 64;
  v8f acc[4] = {};
  for (int k0 = 0; k0 < E_DIM; k0 += 32) {
    v16bf a = load_a_frag(x + (size_t)m0 * E_DIM + k0, E_DIM, lane);
#pragma unroll
    for (int nt = 0; nt < 4; ++nt) {
      v16bf bfr = load_b_frag(w + (size_t)(n0 + nt * 16) * E_DIM + k0, E_DIM, lane);
      acc[nt] = wmma_bf16(a, bfr, acc[nt]);
    }
  }
  const int half = lane >> 4, c = lane & 15;
#pragma unroll
  for (int nt = 0; nt < 4; ++nt) {
    const int n = n0 + nt * 16 + c;
    const float bv = bih[n];
#pragma unroll
    for (int i = 0; i < 8; ++i) {
      const int m = m0 + 8 * half + i;
      xp[(size_t)m * G_DIM + n] = acc[nt][i] + bv;
    }
  }
}

// ---- LSTM recurrence: 4 blocks = {dir0,dir1} x {batch 0-15, 16-31} ---------
// 1024 threads (32 waves). h (bf16, WMMA-A layout source) in LDS; per-step
// g = h @ whh^T via WMMA, gates staged through LDS, c kept in registers.
__global__ void lstm_kernel(const float* __restrict__ xp_f, const float* __restrict__ xp_b,
                            const bf16* __restrict__ whh_f, const bf16* __restrict__ whh_b,
                            const float* __restrict__ bhh_f, const float* __restrict__ bhh_b,
                            float* __restrict__ h_all) {
  extern __shared__ char smem[];
  bf16*  h_sh = (bf16*)smem;                       // 16 x HSTR
  float* g_sh = (float*)(smem + 16 * HSTR * 2);    // 16 x GSTR
  const int dir = blockIdx.x >> 1, bg = blockIdx.x & 1;
  const float* xp  = dir ? xp_b  : xp_f;
  const bf16*  whh = dir ? whh_b : whh_f;
  const float* bhh = dir ? bhh_b : bhh_f;
  const int tid = threadIdx.x, lane = tid & 31, wave = tid >> 5;
  const int mrow = lane & 15, half = lane >> 4;
  const int j0 = wave * 16;     // hidden-unit strip for activation phase
  const int n0 = wave * 64;     // gate-column strip for GEMM phase

  for (int i = tid; i < 16 * HSTR; i += 1024) h_sh[i] = (bf16)0.0f;

  float bi[8], bff[8], bgg[8], boo[8], cc[8];
#pragma unroll
  for (int v = 0; v < 8; ++v) {
    const int j = j0 + 8 * half + v;
    bi[v]  = bhh[j];
    bff[v] = bhh[H_DIM + j];
    bgg[v] = bhh[2 * H_DIM + j];
    boo[v] = bhh[3 * H_DIM + j];
    cc[v] = 0.0f;
  }
  __syncthreads();

  for (int tt = 0; tt < SEQ; ++tt) {
    const int t = dir ? (SEQ - 1 - tt) : tt;
    // prefetch this step's x_proj row segment (L2-resident stream)
    __builtin_prefetch(xp + ((size_t)t * BATCH + bg * 16 + mrow) * G_DIM + j0, 0, 1);

    // ---- recurrent GEMM: g(16 x 2048) = h(16 x 512) @ whh^T ----
    v8f acc[4] = {};
    for (int k0 = 0; k0 < H_DIM; k0 += 32) {
      v16bf a = load_a_frag(h_sh + k0, HSTR, lane);
#pragma unroll
      for (int nt = 0; nt < 4; ++nt) {
        v16bf bfr = load_b_frag(whh + (size_t)(n0 + nt * 16) * H_DIM + k0, H_DIM, lane);
        acc[nt] = wmma_bf16(a, bfr, acc[nt]);
      }
    }
#pragma unroll
    for (int nt = 0; nt < 4; ++nt) {
      const int n = n0 + nt * 16 + mrow;
#pragma unroll
      for (int i = 0; i < 8; ++i)
        g_sh[(size_t)(8 * half + i) * GSTR + n] = acc[nt][i];
    }
    __syncthreads();  // g complete; all h reads done

    // ---- gates & state update: lane -> batch=mrow, units j0+8*half+v ----
    const float* xprow = xp + ((size_t)t * BATCH + bg * 16 + mrow) * G_DIM;
    const float* grow  = g_sh + (size_t)mrow * GSTR;
#pragma unroll
    for (int v = 0; v < 8; ++v) {
      const int j = j0 + 8 * half + v;
      const float gi = grow[j]              + xprow[j]              + bi[v];
      const float gf = grow[H_DIM + j]      + xprow[H_DIM + j]      + bff[v];
      const float gg = grow[2 * H_DIM + j]  + xprow[2 * H_DIM + j]  + bgg[v];
      const float go = grow[3 * H_DIM + j]  + xprow[3 * H_DIM + j]  + boo[v];
      cc[v] = sigmoidf_(gf) * cc[v] + sigmoidf_(gi) * tanhf(gg);
      const float hv = sigmoidf_(go) * tanhf(cc[v]);
      h_sh[(size_t)mrow * HSTR + j] = (bf16)hv;
      h_all[(((size_t)(bg * 16 + mrow)) * SEQ + t) * (2 * H_DIM) + dir * H_DIM + j] = hv;
    }
    __syncthreads();  // new h visible before next step's GEMM
  }
}

// ---- emissions = h_all(B,S,1024) @ w_out^T(25,1024) + b_out ----------------
__global__ void outproj_kernel(const float* __restrict__ h_all, const float* __restrict__ w_out,
                               const float* __restrict__ b_out, float* __restrict__ em) {
  const int idx = blockIdx.x * blockDim.x + threadIdx.x;
  if (idx >= ROWS * T_TAGS) return;
  const int t = idx % T_TAGS, bs = idx / T_TAGS;
  const float* h = h_all + (size_t)bs * (2 * H_DIM);
  const float* w = w_out + (size_t)t * (2 * H_DIM);
  float s = b_out[t];
#pragma unroll 8
  for (int k = 0; k < 2 * H_DIM; ++k) s = fmaf(h[k], w[k], s);
  em[idx] = s;
}

// ---- CRF NLL: 32 waves, wave b handles batch b; lane j<25 holds score[j] ---
__global__ void crf_kernel(const float* __restrict__ em, const unsigned char* __restrict__ mask,
                           const int* __restrict__ tags, const float* __restrict__ st,
                           const float* __restrict__ et, const float* __restrict__ trans,
                           float* __restrict__ out) {
  __shared__ float part[32];
  const int lane = threadIdx.x & 31, b = threadIdx.x >> 5;
  const float* emb_ = em + (size_t)b * SEQ * T_TAGS;
  const unsigned char* mk = mask + (size_t)b * SEQ;
  const int* tg = tags + (size_t)b * SEQ;
  const int jc = (lane < T_TAGS) ? lane : 0;
  float score = (lane < T_TAGS) ? (st[lane] + emb_[lane]) : -3.4e38f;

  for (int s = 1; s < SEQ; ++s) {
    const bool m = (mk[s] != 0);
    float vv[T_TAGS], vmax = -3.4e38f;
#pragma unroll
    for (int i = 0; i < T_TAGS; ++i) {
      const float si = __shfl(score, i, 32);
      vv[i] = si + trans[i * T_TAGS + jc];
      vmax = fmaxf(vmax, vv[i]);
    }
    float ssum = 0.0f;
#pragma unroll
    for (int i = 0; i < T_TAGS; ++i) ssum += __expf(vv[i] - vmax);
    const float nxt = vmax + __logf(ssum) + emb_[s * T_TAGS + jc];
    if (m && lane < T_TAGS) score = nxt;
  }
  // logZ = logsumexp(score + end_trans)
  float z = (lane < T_TAGS) ? (score + et[lane]) : -3.4e38f;
  float zmax = z;
  for (int off = 16; off; off >>= 1) zmax = fmaxf(zmax, __shfl_xor(zmax, off, 32));
  float ze = (lane < T_TAGS) ? __expf(z - zmax) : 0.0f;
  for (int off = 16; off; off >>= 1) ze += __shfl_xor(ze, off, 32);
  const float logZ = zmax + __logf(ze);
  // numerator (lane-parallel over s)
  float np = 0.0f; int cnt = 0;
  for (int s = lane; s < SEQ; s += 32) cnt += (mk[s] != 0);
  for (int s = 1 + lane; s < SEQ; s += 32)
    if (mk[s]) np += trans[tg[s - 1] * T_TAGS + tg[s]] + emb_[s * T_TAGS + tg[s]];
  for (int off = 16; off; off >>= 1) {
    np  += __shfl_xor(np, off, 32);
    cnt += __shfl_xor(cnt, off, 32);
  }
  const float num = np + st[tg[0]] + emb_[tg[0]] + et[tg[cnt - 1]];
  if (lane == 0) part[b] = logZ - num;
  __syncthreads();
  if (threadIdx.x == 0) {
    float tot = 0.0f;
    for (int i = 0; i < 32; ++i) tot += part[i];
    out[0] = tot;   // -(num - logZ).sum()
  }
}

// ---------------------------------------------------------------------------
extern "C" void kernel_launch(void* const* d_in, const int* in_sizes, int n_in,
                              void* d_out, int out_size, void* d_ws, size_t ws_size,
                              hipStream_t stream) {
  const float* embed   = (const float*)d_in[0];
  const float* w_ih_f  = (const float*)d_in[1];
  const float* w_hh_f  = (const float*)d_in[2];
  const float* b_ih_f  = (const float*)d_in[3];
  const float* b_hh_f  = (const float*)d_in[4];
  const float* w_ih_b  = (const float*)d_in[5];
  const float* w_hh_b  = (const float*)d_in[6];
  const float* b_ih_b  = (const float*)d_in[7];
  const float* b_hh_b  = (const float*)d_in[8];
  const float* w_out   = (const float*)d_in[9];
  const float* b_out   = (const float*)d_in[10];
  const float* s_tr    = (const float*)d_in[11];
  const float* e_tr    = (const float*)d_in[12];
  const float* trans   = (const float*)d_in[13];
  const int*   input_ix = (const int*)d_in[14];
  const unsigned char* amask = (const unsigned char*)d_in[15];
  const int*   labels  = (const int*)d_in[16];
  float* out = (float*)d_out;

  char* ws = (char*)d_ws;
  size_t off = 0;
  auto alloc = [&](size_t bytes) -> void* {
    void* p = ws + off;
    off += (bytes + 255) & ~(size_t)255;
    return p;
  };
  bf16*  x_bf   = (bf16*) alloc((size_t)ROWS * E_DIM * 2);       // 4 MB
  bf16*  wih_f  = (bf16*) alloc((size_t)G_DIM * E_DIM * 2);      // 1 MB
  bf16*  wih_b  = (bf16*) alloc((size_t)G_DIM * E_DIM * 2);
  bf16*  whh_f  = (bf16*) alloc((size_t)G_DIM * H_DIM * 2);      // 2 MB
  bf16*  whh_b  = (bf16*) alloc((size_t)G_DIM * H_DIM * 2);
  float* xp_f   = (float*)alloc((size_t)ROWS * G_DIM * 4);       // 64 MB
  float* xp_b   = (float*)alloc((size_t)ROWS * G_DIM * 4);
  float* h_all  = (float*)alloc((size_t)ROWS * 2 * H_DIM * 4);   // 32 MB
  float* emis   = (float*)alloc((size_t)ROWS * T_TAGS * 4);      // 0.8 MB

  cvt_bf16_kernel<<<512, 256, 0, stream>>>(w_ih_f, wih_f, G_DIM * E_DIM);
  cvt_bf16_kernel<<<512, 256, 0, stream>>>(w_ih_b, wih_b, G_DIM * E_DIM);
  cvt_bf16_kernel<<<1024, 256, 0, stream>>>(w_hh_f, whh_f, G_DIM * H_DIM);
  cvt_bf16_kernel<<<1024, 256, 0, stream>>>(w_hh_b, whh_b, G_DIM * H_DIM);

  gather_kernel<<<(ROWS * E_DIM) / 256, 256, 0, stream>>>(embed, input_ix, x_bf);

  inproj_kernel<<<dim3(G_DIM / 512, ROWS / 16, 2), 256, 0, stream>>>(
      x_bf, wih_f, wih_b, b_ih_f, b_ih_b, xp_f, xp_b);

  hipFuncSetAttribute((const void*)lstm_kernel,
                      hipFuncAttributeMaxDynamicSharedMemorySize, LSTM_LDS_BYTES);
  lstm_kernel<<<4, 1024, LSTM_LDS_BYTES, stream>>>(
      xp_f, xp_b, whh_f, whh_b, b_hh_f, b_hh_b, h_all);

  outproj_kernel<<<(ROWS * T_TAGS + 255) / 256, 256, 0, stream>>>(h_all, w_out, b_out, emis);

  crf_kernel<<<1, 1024, 0, stream>>>(emis, amask, labels, s_tr, e_tr, trans, out);
}
